// Attention_75204877353110
// MI455X (gfx1250) — compile-verified, matrix operand order checked by reading, not run on hardware
//
#include <hip/hip_runtime.h>
#include <hip/hip_bf16.h>
#include <math.h>
#include <stdint.h>

// ---------------------------------------------------------------------------
// Problem constants (match reference)
// ---------------------------------------------------------------------------
#define S_LEN   2048
#define D_MODEL 2048
#define N_HEADS 32
#define N_KV    8
#define HEAD_D  64
#define WINDOW  1024
#define QH_DIM  (N_HEADS * HEAD_D)   // 2048
#define KV_DIM  (N_KV * HEAD_D)      // 512

typedef __attribute__((ext_vector_type(16))) __bf16 v16bf;
typedef __attribute__((ext_vector_type(8)))  __bf16 v8bf;
typedef __attribute__((ext_vector_type(8)))  float  v8f;

// ---------------------------------------------------------------------------
// WMMA helper (bf16 inputs, f32 accumulate) — wave32, 16x16x32
// ---------------------------------------------------------------------------
__device__ __forceinline__ v8f wmma_bf16(v16bf a, v16bf b, v8f c) {
    return __builtin_amdgcn_wmma_f32_16x16x32_bf16(
        /*neg_a=*/false, a, /*neg_b=*/false, b,
        /*c_mod=*/(short)0, c, /*reuse_a=*/false, /*reuse_b=*/false);
}

// Fragment load: per-lane 16 bf16 as two contiguous 16-byte chunks
// (elements [p..p+7] and [p+16..p+23]); p must be 16B aligned.
__device__ __forceinline__ v16bf load_frag(const __bf16* p) {
    v8bf lo = *reinterpret_cast<const v8bf*>(p);
    v8bf hi = *reinterpret_cast<const v8bf*>(p + 16);
    return __builtin_shufflevector(lo, hi, 0, 1, 2, 3, 4, 5, 6, 7,
                                   8, 9, 10, 11, 12, 13, 14, 15);
}

// CDNA5 async copy: 16 bytes global -> LDS per enabled lane (ASYNCcnt).
__device__ __forceinline__ void async_load_b128(uint32_t lds_off,
                                                const void* gaddr) {
    asm volatile("global_load_async_to_lds_b128 %0, %1, off"
                 :: "v"(lds_off), "v"(gaddr) : "memory");
}
__device__ __forceinline__ void wait_async0() {
    asm volatile("s_wait_asynccnt 0x0" ::: "memory");
}

// ---------------------------------------------------------------------------
// GEMM: C[M,N] = A[M,K] @ B[K,N]; A bf16 row-major [M,K], B pre-transposed
// bf16 [N,K]. 4 waves / 128 threads per workgroup, 128x64 C tile per WG
// (each wave: 32x64 = 8 WMMA accumulators). The shared 64x32 B tile is
// staged through double-buffered LDS with global_load_async_to_lds_b128,
// overlapped with compute and fenced by s_wait_asynccnt + barrier.
// grid.x = (M/128) * (N/64)
// ---------------------------------------------------------------------------
__global__ void gemm_bf16_wmma(const __bf16* __restrict__ A,
                               const __bf16* __restrict__ Bt,
                               float* __restrict__ C,
                               int M, int N, int K) {
    __shared__ __align__(16) __bf16 bsh[2][64 * 32];   // [buf][col*32 + k]
    const uint32_t BUFSZ = 64 * 32 * 2;                // 4096 bytes

    int ntn   = N >> 6;
    int wg_m0 = (blockIdx.x / ntn) << 7;               // 128 rows per WG
    int n0    = (blockIdx.x % ntn) << 6;               // 64 cols
    int tid   = threadIdx.x;
    int lane  = tid & 31;
    int wave  = tid >> 5;
    int m0    = wg_m0 + (wave << 5);                   // 32 rows per wave
    int lrow  = lane & 15;
    int koff  = (lane >> 4) << 3;                      // 0 or 8

    // async staging: 2 x 16B chunks per thread cover the 4KB B tile
    int c0 = tid, c1 = tid + 128;
    int col0 = c0 >> 2, kp0 = (c0 & 3) << 3;
    int col1 = c1 >> 2, kp1 = (c1 & 3) << 3;
    const __bf16* gb0 = Bt + (size_t)(n0 + col0) * K + kp0;
    const __bf16* gb1 = Bt + (size_t)(n0 + col1) * K + kp1;
    uint32_t l0 = (uint32_t)(uintptr_t)&bsh[0][col0 * 32 + kp0];
    uint32_t l1 = (uint32_t)(uintptr_t)&bsh[0][col1 * 32 + kp1];

    const __bf16* pa0 = A + (size_t)(m0 + lrow) * K + koff;
    const __bf16* pa1 = A + (size_t)(m0 + 16 + lrow) * K + koff;

    // prime buffer 0
    async_load_b128(l0, gb0);
    async_load_b128(l1, gb1);
    wait_async0();
    __syncthreads();

    v8f acc[2][4] = {};
    int buf = 0;
    for (int k0 = 0; k0 < K; k0 += 32) {
        int nbuf = buf ^ 1;
        if (k0 + 32 < K) {                              // stage next tile
            async_load_b128(l0 + nbuf * BUFSZ, gb0 + k0 + 32);
            async_load_b128(l1 + nbuf * BUFSZ, gb1 + k0 + 32);
            __builtin_prefetch(pa0 + k0 + 32, 0, 3);    // near-cache prefetch
            __builtin_prefetch(pa1 + k0 + 32, 0, 3);
        }
        v16bf a0 = load_frag(pa0 + k0);
        v16bf a1 = load_frag(pa1 + k0);
        const __bf16* bb = &bsh[buf][0];
#pragma unroll
        for (int c = 0; c < 4; ++c) {
            v16bf b = load_frag(bb + (16 * c + lrow) * 32 + koff);
            acc[0][c] = wmma_bf16(a0, b, acc[0][c]);
            acc[1][c] = wmma_bf16(a1, b, acc[1][c]);
        }
        if (k0 + 32 < K) wait_async0();                 // own async loads done
        __syncthreads();                                // tile visible to WG
        buf = nbuf;
    }

    int col = lane & 15;
    int mhi = (lane >> 4) << 3;                         // 0 or 8
#pragma unroll
    for (int s = 0; s < 2; ++s)
#pragma unroll
        for (int c = 0; c < 4; ++c)
#pragma unroll
            for (int r = 0; r < 8; ++r)
                C[(size_t)(m0 + 16 * s + r + mhi) * N + n0 + 16 * c + col] =
                    acc[s][c][r];
}

// ---------------------------------------------------------------------------
// Packing kernels
// ---------------------------------------------------------------------------
// f32 [K,N] -> bf16 transposed [N,K] (output-coalesced)
__global__ void pack_wt_bf16(const float* __restrict__ w,
                             __bf16* __restrict__ wt, int K, int N) {
    size_t o = (size_t)blockIdx.x * blockDim.x + threadIdx.x;
    if (o >= (size_t)K * N) return;
    int k = (int)(o % K);
    int n = (int)(o / K);
    wt[o] = (__bf16)w[(size_t)k * N + n];
}

// f32 [S, KV_DIM] -> bf16 transposed [KV_DIM, S]
__global__ void pack_vt_bf16(const float* __restrict__ v,
                             __bf16* __restrict__ vt) {
    size_t o = (size_t)blockIdx.x * blockDim.x + threadIdx.x;
    if (o >= (size_t)S_LEN * KV_DIM) return;
    int s = (int)(o % S_LEN);
    int d = (int)(o / S_LEN);
    vt[o] = (__bf16)v[(size_t)s * KV_DIM + d];
}

__global__ void cvt_bf16(const float* __restrict__ src,
                         __bf16* __restrict__ dst, int n) {
    int idx = blockIdx.x * blockDim.x + threadIdx.x;
    if (idx < n) dst[idx] = (__bf16)src[idx];
}

// RoPE + f32 -> bf16 pack. One thread per (s, head, pair).
__global__ void rope_pack_bf16(const float* __restrict__ src,
                               const float* __restrict__ cs,
                               const float* __restrict__ sn,
                               __bf16* __restrict__ dst, int nheads) {
    int idx   = blockIdx.x * blockDim.x + threadIdx.x;
    int total = S_LEN * nheads * (HEAD_D / 2);
    if (idx >= total) return;
    int pair = idx & 31;
    int t    = idx >> 5;
    int h    = t % nheads;
    int s    = t / nheads;
    float c  = cs[s * 32 + pair];
    float si = sn[s * 32 + pair];
    size_t base = ((size_t)s * nheads + h) * HEAD_D + pair * 2;
    float a = src[base], b = src[base + 1];
    dst[base]     = (__bf16)(a * c - b * si);
    dst[base + 1] = (__bf16)(a * si + b * c);
}

// ---------------------------------------------------------------------------
// Flash attention, sliding-window causal, GQA (4 Q heads per KV head).
// One wave per (query-tile of 16, head). 32 keys per inner iteration:
// 4 score WMMAs + 4 P@V WMMAs. Vt is [KV_DIM, S] (keys contiguous).
// ---------------------------------------------------------------------------
__global__ void attn_wmma(const __bf16* __restrict__ Q,
                          const __bf16* __restrict__ Kc,
                          const __bf16* __restrict__ Vt,
                          float* __restrict__ O) {
    __shared__ float  s_s[16][32];
    __shared__ __align__(16) __bf16 s_p[16][32];
    __shared__ float  s_alpha[16];
    __shared__ float  s_l[16];

    int q0   = blockIdx.x << 4;
    int h    = blockIdx.y;
    int kvh  = h >> 2;                  // H / KV = 4
    int lane = threadIdx.x & 31;
    int lrow = lane & 15;
    int koff = (lane >> 4) << 3;        // 0 or 8
    int mhi  = koff;

    // Q fragments for head-dim chunks kc=0,1 (each 16x32)
    v16bf qf[2];
#pragma unroll
    for (int kc = 0; kc < 2; ++kc)
        qf[kc] = load_frag(Q + (size_t)(q0 + lrow) * QH_DIM +
                           h * HEAD_D + kc * 32 + koff);

    v8f  acc[4] = {};
    float m_i = -1e30f, l_i = 0.0f;     // live in lanes 0..15

    int kb0 = q0 > WINDOW ? ((q0 - WINDOW) & ~31) : 0;
    for (int kb = kb0; kb <= q0 + 15; kb += 32) {
        // ---- scores: two 16x16 tiles over keys kb..kb+31 ----
        v8f st[2] = {};
#pragma unroll
        for (int t = 0; t < 2; ++t) {
            int key = kb + 16 * t + lrow;
#pragma unroll
            for (int kc = 0; kc < 2; ++kc) {
                v16bf kf = load_frag(Kc + (size_t)key * KV_DIM +
                                     kvh * HEAD_D + kc * 32 + koff);
                st[t] = wmma_bf16(qf[kc], kf, st[t]);
            }
        }
        // ---- scale + mask, dump to LDS ----
#pragma unroll
        for (int t = 0; t < 2; ++t)
#pragma unroll
            for (int r = 0; r < 8; ++r) {
                int qi = q0 + r + mhi;
                int kj = kb + 16 * t + lrow;
                float v = st[t][r] * 0.125f;   // 1/sqrt(64)
                bool ok = (kj <= qi) && (kj >= qi - WINDOW);
                s_s[r + mhi][16 * t + lrow] = ok ? v : -1e30f;
            }
        __syncthreads();

        // ---- online softmax: lane r owns query row r ----
        if (lane < 16) {
            float mx = m_i;
#pragma unroll
            for (int j = 0; j < 32; ++j) mx = fmaxf(mx, s_s[lane][j]);
            float alpha = __expf(m_i - mx);
            float sum = 0.0f;
#pragma unroll
            for (int j = 0; j < 32; ++j) {
                float p = __expf(s_s[lane][j] - mx);
                sum += p;
                s_p[lane][j] = (__bf16)p;
            }
            l_i = l_i * alpha + sum;
            m_i = mx;
            s_alpha[lane] = alpha;
        }
        __syncthreads();

        // ---- rescale accumulator, then P @ V ----
#pragma unroll
        for (int c = 0; c < 4; ++c)
#pragma unroll
            for (int r = 0; r < 8; ++r) acc[c][r] *= s_alpha[r + mhi];

        v16bf pf = load_frag(&s_p[lrow][koff]);   // A-frag of P (16 x 32 keys)

#pragma unroll
        for (int c = 0; c < 4; ++c) {    // 4 chunks of 16 head-dim columns
            int hd = kvh * HEAD_D + c * 16 + lrow;
            v16bf vf = load_frag(Vt + (size_t)hd * S_LEN + kb + koff);
            acc[c] = wmma_bf16(pf, vf, acc[c]);
        }
        __syncthreads();
    }

    if (lane < 16) s_l[lane] = l_i;
    __syncthreads();

#pragma unroll
    for (int c = 0; c < 4; ++c)
#pragma unroll
        for (int r = 0; r < 8; ++r) {
            int qi = q0 + r + mhi;
            O[(size_t)qi * QH_DIM + h * HEAD_D + c * 16 + lrow] =
                acc[c][r] / s_l[r + mhi];
        }
}

// ---------------------------------------------------------------------------
// Host-side launch
// ---------------------------------------------------------------------------
extern "C" void kernel_launch(void* const* d_in, const int* in_sizes, int n_in,
                              void* d_out, int out_size, void* d_ws, size_t ws_size,
                              hipStream_t stream) {
    const float* x   = (const float*)d_in[0];
    const float* wq  = (const float*)d_in[1];
    const float* wk  = (const float*)d_in[2];
    const float* wv  = (const float*)d_in[3];
    const float* wo  = (const float*)d_in[4];
    const float* fcs = (const float*)d_in[5];
    const float* fsn = (const float*)d_in[6];
    float* out = (float*)d_out;

    // ---- workspace layout (64 MB, with deliberate aliasing) ----
    char* ws = (char*)d_ws;
    float*  q_f32   = (float*)ws;                 // 16 MB
    float*  att_f32 = q_f32;                      // alias: q_f32 dead after RoPE
    ws += (size_t)S_LEN * QH_DIM * 4;
    float*  k_f32 = (float*)ws;  ws += (size_t)S_LEN * KV_DIM * 4;   // 4 MB
    float*  v_f32 = (float*)ws;  ws += (size_t)S_LEN * KV_DIM * 4;   // 4 MB
    __bf16* x_bf   = (__bf16*)ws;                 // 8 MB
    __bf16* att_bf = x_bf;                        // alias: x_bf dead after QKV
    ws += (size_t)S_LEN * D_MODEL * 2;
    __bf16* wq_t  = (__bf16*)ws; ws += (size_t)D_MODEL * QH_DIM * 2; // 8 MB
    __bf16* wk_t  = (__bf16*)ws; ws += (size_t)D_MODEL * KV_DIM * 2; // 2 MB
    __bf16* wv_t  = (__bf16*)ws; ws += (size_t)D_MODEL * KV_DIM * 2; // 2 MB
    __bf16* wo_t  = (__bf16*)ws; ws += (size_t)QH_DIM * D_MODEL * 2; // 8 MB
    __bf16* q_bf  = (__bf16*)ws; ws += (size_t)S_LEN * QH_DIM * 2;   // 8 MB
    __bf16* k_bf  = (__bf16*)ws; ws += (size_t)S_LEN * KV_DIM * 2;   // 2 MB
    __bf16* v_bft = (__bf16*)ws; ws += (size_t)S_LEN * KV_DIM * 2;   // 2 MB

    // ---- 0) pack inputs to bf16 (weights transposed to [N,K]) ----
    {
        int nd = S_LEN * D_MODEL;
        cvt_bf16<<<(nd + 255) / 256, 256, 0, stream>>>(x, x_bf, nd);
        int nq = D_MODEL * QH_DIM;
        pack_wt_bf16<<<(nq + 255) / 256, 256, 0, stream>>>(wq, wq_t, D_MODEL, QH_DIM);
        int nk = D_MODEL * KV_DIM;
        pack_wt_bf16<<<(nk + 255) / 256, 256, 0, stream>>>(wk, wk_t, D_MODEL, KV_DIM);
        pack_wt_bf16<<<(nk + 255) / 256, 256, 0, stream>>>(wv, wv_t, D_MODEL, KV_DIM);
        int no = QH_DIM * D_MODEL;
        pack_wt_bf16<<<(no + 255) / 256, 256, 0, stream>>>(wo, wo_t, QH_DIM, D_MODEL);
    }

    // ---- 1) projections (bf16 WMMA GEMMs, async-LDS staged) ----
    gemm_bf16_wmma<<<(S_LEN / 128) * (QH_DIM / 64), 128, 0, stream>>>(
        x_bf, wq_t, q_f32, S_LEN, QH_DIM, D_MODEL);
    gemm_bf16_wmma<<<(S_LEN / 128) * (KV_DIM / 64), 128, 0, stream>>>(
        x_bf, wk_t, k_f32, S_LEN, KV_DIM, D_MODEL);
    gemm_bf16_wmma<<<(S_LEN / 128) * (KV_DIM / 64), 128, 0, stream>>>(
        x_bf, wv_t, v_f32, S_LEN, KV_DIM, D_MODEL);

    // ---- 2) RoPE + bf16 pack (V transposed for P@V fragments) ----
    {
        int nq = S_LEN * N_HEADS * (HEAD_D / 2);
        rope_pack_bf16<<<(nq + 255) / 256, 256, 0, stream>>>(q_f32, fcs, fsn, q_bf, N_HEADS);
        int nk = S_LEN * N_KV * (HEAD_D / 2);
        rope_pack_bf16<<<(nk + 255) / 256, 256, 0, stream>>>(k_f32, fcs, fsn, k_bf, N_KV);
        int nv = S_LEN * KV_DIM;
        pack_vt_bf16<<<(nv + 255) / 256, 256, 0, stream>>>(v_f32, v_bft);
    }

    // ---- 3) sliding-window flash attention ----
    attn_wmma<<<dim3(S_LEN / 16, N_HEADS), 32, 0, stream>>>(q_bf, k_bf, v_bft, att_f32);

    // ---- 4) output projection ----
    {
        int na = S_LEN * QH_DIM;
        cvt_bf16<<<(na + 255) / 256, 256, 0, stream>>>(att_f32, att_bf, na);
        gemm_bf16_wmma<<<(S_LEN / 128) * (D_MODEL / 64), 128, 0, stream>>>(
            att_bf, wo_t, out, S_LEN, D_MODEL, D_MODEL);
    }
}
